// ModelEdge_57277683860071
// MI455X (gfx1250) — compile-verified, matrix operand order checked by reading, not run on hardware
//
#include <hip/hip_runtime.h>
#include <hip/hip_bf16.h>

// ---------------- problem constants ----------------
#define NN    100000
#define EE    20000
#define NNZK  500000
#define CDIM  256
#define ODIM  128
#define NEGS  0.01f

typedef __attribute__((ext_vector_type(16))) __bf16 bf16x16;
typedef __attribute__((ext_vector_type(8)))  float  f32x8;

enum { FLEAKY = 1, FACC = 2, FBIAS = 4 };

// ---------------- device helpers ----------------
__device__ __forceinline__ unsigned short f2bf(float f) {
  unsigned u = __float_as_uint(f);
  unsigned r = u + 0x7FFFu + ((u >> 16) & 1u);   // RNE
  return (unsigned short)(r >> 16);
}

__device__ __forceinline__ float wred(float s) {
#pragma unroll
  for (int m = 16; m; m >>= 1) s += __shfl_xor(s, m, 32);
  return s;
}

// hardware float atomic-min (GLOBAL_ATOMIC_MIN_NUM_F32)
__device__ __forceinline__ void atomicMinF(float* addr, float val) {
  __hip_atomic_fetch_min(addr, val, __ATOMIC_RELAXED, __HIP_MEMORY_SCOPE_AGENT);
}

// ---------------- elementwise / utility kernels ----------------
__global__ void k_fill(float* p, long long n, float v) {
  long long i = blockIdx.x * (long long)blockDim.x + threadIdx.x;
  long long st = (long long)gridDim.x * blockDim.x;
  for (; i < n; i += st) p[i] = v;
}

// convert f32 -> bf16 (rows x cols, source leading dim ld)
__global__ void k_cvt_bf16(unsigned short* __restrict__ dst, const float* __restrict__ src,
                           long long rows, long long cols, long long ld) {
  long long tot = rows * cols;
  long long i = blockIdx.x * (long long)blockDim.x + threadIdx.x;
  long long st = (long long)gridDim.x * blockDim.x;
  for (; i < tot; i += st) {
    long long r = i / cols, c = i - r * cols;
    dst[i] = f2bf(src[r * ld + c]);
  }
}

// per-row (optional L2-normalize) + LayerNorm -> bf16  (one wave per 256-wide row)
__global__ void k_ln_bf16(const float* __restrict__ x, const float* __restrict__ w,
                          const float* __restrict__ b, unsigned short* __restrict__ dst,
                          int rows, int l2) {
  int wid  = (int)((blockIdx.x * (long long)blockDim.x + threadIdx.x) >> 5);
  int lane = threadIdx.x & 31;
  if (wid >= rows) return;
  const float* px = x + (long long)wid * CDIM;
  int c0 = lane * 8;
  float v[8];
#pragma unroll
  for (int j = 0; j < 8; ++j) v[j] = px[c0 + j];
  if (l2) {
    float ss = 0.f;
#pragma unroll
    for (int j = 0; j < 8; ++j) ss += v[j] * v[j];
    ss = wred(ss);
    float inv = 1.f / fmaxf(sqrtf(ss), 1e-12f);
#pragma unroll
    for (int j = 0; j < 8; ++j) v[j] *= inv;
  }
  float s = 0.f;
#pragma unroll
  for (int j = 0; j < 8; ++j) s += v[j];
  float mu = wred(s) * (1.f / CDIM);
  float q = 0.f;
#pragma unroll
  for (int j = 0; j < 8; ++j) { float d = v[j] - mu; q += d * d; }
  float rstd = rsqrtf(wred(q) * (1.f / CDIM) + 1e-5f);
  unsigned short* pd = dst + (long long)wid * CDIM + c0;
#pragma unroll
  for (int j = 0; j < 8; ++j) pd[j] = f2bf((v[j] - mu) * rstd * w[c0 + j] + b[c0 + j]);
}

// column sum (one thread per column, block handles 256-row slab)
__global__ void k_colsum(const float* __restrict__ x, float* acc, int rows) {
  int c = threadIdx.x;
  int r0 = blockIdx.x * 256;
  int r1 = r0 + 256; if (r1 > rows) r1 = rows;
  float s = 0.f;
  for (int r = r0; r < r1; ++r) s += x[(long long)r * CDIM + c];
  atomicAdd(&acc[c], s);
}

// column variance of (x - ms*mean)
__global__ void k_colvar(const float* __restrict__ x, const float* __restrict__ cm,
                         const float* __restrict__ ms, float* cv, int rows, float invR) {
  int c = threadIdx.x;
  float sub = ms[c] * (cm[c] * invR);
  int r0 = blockIdx.x * 256;
  int r1 = r0 + 256; if (r1 > rows) r1 = rows;
  float s = 0.f;
  for (int r = r0; r < r1; ++r) { float v = x[(long long)r * CDIM + c] - sub; s += v * v; }
  atomicAdd(&cv[c], s);
}

// GraphNorm apply -> f32 + bf16
__global__ void k_gn_apply(const float* __restrict__ x, float* __restrict__ of,
                           unsigned short* __restrict__ ob,
                           const float* __restrict__ cm, const float* __restrict__ cv,
                           const float* __restrict__ w, const float* __restrict__ b,
                           const float* __restrict__ ms, long long total, float invR) {
  long long i = blockIdx.x * (long long)blockDim.x + threadIdx.x;
  long long st = (long long)gridDim.x * blockDim.x;
  for (; i < total; i += st) {
    int c = (int)(i & (CDIM - 1));
    float mean = cm[c] * invR;
    float var  = cv[c] * invR;
    float o = x[i] - ms[c] * mean;
    float y = w[c] * o * rsqrtf(var + 1e-5f) + b[c];
    of[i] = y;
    ob[i] = f2bf(y);
  }
}

// incidence degrees
__global__ void k_degree(const int* __restrict__ row, const int* __restrict__ col,
                         float* D, float* B, int nnz) {
  long long i = blockIdx.x * (long long)blockDim.x + threadIdx.x;
  long long st = (long long)gridDim.x * blockDim.x;
  for (; i < nnz; i += st) {
    atomicAdd(&D[row[i]], 1.f);
    atomicAdd(&B[col[i]], 1.f);
  }
}

// dst[di[k]] += scale(si[k]) * src[si[k]] over 256-wide rows.
// 64 threads (float4 each) per entry; scnt (nullable) folds the 1/deg scaling of the
// source rows into the gather read, saving a whole pass over the intermediate tensor.
__global__ void k_scatter_add(float* __restrict__ dst, const float* __restrict__ src,
                              const int* __restrict__ di, const int* __restrict__ si,
                              const float* __restrict__ scnt, int nnz) {
  long long total = (long long)nnz * 64;
  long long t = blockIdx.x * (long long)blockDim.x + threadIdx.x;
  long long st = (long long)gridDim.x * blockDim.x;
  for (; t < total; t += st) {
    int k  = (int)(t >> 6);
    int c4 = (int)((t & 63) << 2);
    int sidx = si[k];
    if (c4 == 0 && k + 4 < nnz)   // hide index->row gather latency
      __builtin_prefetch(src + (long long)si[k + 4] * CDIM, 0, 0);
    float sc = 1.f;
    if (scnt) { float c = scnt[sidx]; sc = c > 0.f ? 1.f / c : 0.f; }
    float4 v = *reinterpret_cast<const float4*>(src + (long long)sidx * CDIM + c4);
    float* p = dst + (long long)di[k] * CDIM + c4;
    atomicAdd(p + 0, v.x * sc);
    atomicAdd(p + 1, v.y * sc);
    atomicAdd(p + 2, v.z * sc);
    atomicAdd(p + 3, v.w * sc);
  }
}

// segment-min scatter (hardware float min atomics)
__global__ void k_scatter_min(float* __restrict__ dst, const float* __restrict__ src,
                              const int* __restrict__ di, const int* __restrict__ si, int nnz) {
  long long total = (long long)nnz * 64;
  long long t = blockIdx.x * (long long)blockDim.x + threadIdx.x;
  long long st = (long long)gridDim.x * blockDim.x;
  for (; t < total; t += st) {
    int k  = (int)(t >> 6);
    int c4 = (int)((t & 63) << 2);
    int sidx = si[k];
    if (c4 == 0 && k + 4 < nnz)
      __builtin_prefetch(src + (long long)si[k + 4] * CDIM, 0, 0);
    float4 v = *reinterpret_cast<const float4*>(src + (long long)sidx * CDIM + c4);
    float* p = dst + (long long)di[k] * CDIM + c4;
    atomicMinF(p + 0, v.x);
    atomicMinF(p + 1, v.y);
    atomicMinF(p + 2, v.z);
    atomicMinF(p + 3, v.w);
  }
}

// x = leaky(x/cnt + bias)
__global__ void k_scale_bias_leaky(float* __restrict__ x, const float* __restrict__ cnt,
                                   const float* __restrict__ bias, long long total) {
  long long i = blockIdx.x * (long long)blockDim.x + threadIdx.x;
  long long st = (long long)gridDim.x * blockDim.x;
  for (; i < total; i += st) {
    float c = cnt[i >> 8];
    float inv = c > 0.f ? 1.f / c : 0.f;
    float v = x[i] * inv + bias[i & (CDIM - 1)];
    x[i] = v >= 0.f ? v : NEGS * v;
  }
}

// replace non-finite with 0 (empty segment-min -> 0)
__global__ void k_fixfinite(float* __restrict__ x, long long total) {
  long long i = blockIdx.x * (long long)blockDim.x + threadIdx.x;
  long long st = (long long)gridDim.x * blockDim.x;
  for (; i < total; i += st) {
    float v = x[i];
    x[i] = __builtin_isfinite(v) ? v : 0.f;
  }
}

// ---------------- WMMA GEMM: C[M,Nn] (+)= leaky( A[M,K]bf16 * W[Nn,K]bf16^T + bias ) ------
// one wave per 16x16 tile; fragments follow the CDNA5 16-bit A layout
// (lane<16 -> K chunks {0..7,16..23}, lane>=16 -> {8..15,24..31}), B symmetric via W row-major.
__global__ void k_gemm(const unsigned short* __restrict__ A, const unsigned short* __restrict__ W,
                       const float* __restrict__ bias, float* __restrict__ C,
                       int M, int Nn, int K, int flags) {
  int tid = blockIdx.x * (blockDim.x >> 5) + (threadIdx.x >> 5);
  int tilesN = Nn >> 4;
  int bm = tid / tilesN;
  int bn = tid - bm * tilesN;
  if (bm >= (M >> 4)) return;
  int lane  = threadIdx.x & 31;
  int mrow  = (bm << 4) + (lane & 15);
  int nrow  = (bn << 4) + (lane & 15);
  int khalf = (lane >> 4) << 3;              // 0 or 8
  const unsigned short* pa = A + (long long)mrow * K + khalf;
  const unsigned short* pw = W + (long long)nrow * K + khalf;
  union FragU { bf16x16 v; uint4 q[2]; };
  f32x8 acc = {};
  for (int kb = 0; kb < K; kb += 32) {
    FragU fa, fb;
    fa.q[0] = *reinterpret_cast<const uint4*>(pa + kb);
    fa.q[1] = *reinterpret_cast<const uint4*>(pa + kb + 16);
    fb.q[0] = *reinterpret_cast<const uint4*>(pw + kb);
    fb.q[1] = *reinterpret_cast<const uint4*>(pw + kb + 16);
    acc = __builtin_amdgcn_wmma_f32_16x16x32_bf16(false, fa.v, false, fb.v,
                                                  (short)0, acc, false, false);
  }
  int orow = (bm << 4) + ((lane >> 4) << 3);
  int ocol = (bn << 4) + (lane & 15);
  float bv = (flags & FBIAS) ? bias[ocol] : 0.f;
#pragma unroll
  for (int r = 0; r < 8; ++r) {
    float v = acc[r] + bv;
    if (flags & FLEAKY) v = v >= 0.f ? v : NEGS * v;
    long long idx = (long long)(orow + r) * Nn + ocol;
    if (flags & FACC) C[idx] += v; else C[idx] = v;
  }
}

// ---------------- host orchestration ----------------
static inline unsigned gs_blocks(long long total) {
  long long b = (total + 255) / 256;
  if (b > 4096) b = 4096;
  if (b < 1) b = 1;
  return (unsigned)b;
}

extern "C" void kernel_launch(void* const* d_in, const int* in_sizes, int n_in,
                              void* d_out, int out_size, void* d_ws, size_t ws_size,
                              hipStream_t stream) {
  const float* x  = (const float*)d_in[0];
  const float* xe = (const float*)d_in[1];
  const int*   ei = (const int*)d_in[2];
  const int* row = ei;
  const int* col = ei + NNZK;

  const float* in_norm_w = (const float*)d_in[3];
  const float* in_norm_b = (const float*)d_in[4];
  const float* in_proj_w = (const float*)d_in[5];
  const float* in_proj_b = (const float*)d_in[6];
  const float* e_norm_w  = (const float*)d_in[7];
  const float* e_norm_b  = (const float*)d_in[8];
  const float* e_proj_w  = (const float*)d_in[9];
  const float* e_proj_b  = (const float*)d_in[10];
  const float* gn0_w     = (const float*)d_in[11];
  const float* gn0_b     = (const float*)d_in[12];
  const float* gn0_ms    = (const float*)d_in[13];
  const float* hg0_w     = (const float*)d_in[14];
  const float* hg0_b     = (const float*)d_in[15];
  const float* skip0_w   = (const float*)d_in[16];
  const float* skip0_b   = (const float*)d_in[17];
  const float* gnd0_w    = (const float*)d_in[18];
  const float* gnd0_b    = (const float*)d_in[19];
  const float* gnd0_ms   = (const float*)d_in[20];
  const float* hgd0_w    = (const float*)d_in[21];
  const float* hgd0_b    = (const float*)d_in[22];
  const float* skipd0_w  = (const float*)d_in[23];
  const float* skipd0_b  = (const float*)d_in[24];
  const float* fuse_w    = (const float*)d_in[25];
  const float* fuse_b    = (const float*)d_in[26];
  const float* lin_w     = (const float*)d_in[27];
  const float* lin_b     = (const float*)d_in[28];

  float* out = (float*)d_out;

  char* ws = (char*)d_ws;
  size_t off = 0;
  auto alloc = [&](size_t bytes) -> void* {
    off = (off + 255) & ~(size_t)255;
    void* p = ws + off;
    off += bytes;
    return p;
  };

  unsigned short* W_in     = (unsigned short*)alloc(65536 * 2);
  unsigned short* W_e      = (unsigned short*)alloc(65536 * 2);
  unsigned short* W_hg0    = (unsigned short*)alloc(65536 * 2);
  unsigned short* W_skip0  = (unsigned short*)alloc(65536 * 2);
  unsigned short* W_hgd0   = (unsigned short*)alloc(65536 * 2);
  unsigned short* W_skipd0 = (unsigned short*)alloc(65536 * 2);
  unsigned short* W_fuse1  = (unsigned short*)alloc(65536 * 2);
  unsigned short* W_fuse2  = (unsigned short*)alloc(65536 * 2);
  unsigned short* W_lin    = (unsigned short*)alloc(32768 * 2);

  unsigned short* nbf = (unsigned short*)alloc((size_t)NN * CDIM * 2);
  unsigned short* ebf = (unsigned short*)alloc((size_t)EE * CDIM * 2);
  float* fA = (float*)alloc((size_t)NN * CDIM * 4);
  float* fB = (float*)alloc((size_t)NN * CDIM * 4);
  float* fC = (float*)alloc((size_t)NN * CDIM * 4);
  float* eA = (float*)alloc((size_t)EE * CDIM * 4);
  float* eB = (float*)alloc((size_t)EE * CDIM * 4);
  float* eC = (float*)alloc((size_t)EE * CDIM * 4);
  float* cm = (float*)alloc(256 * 4);
  float* cv = (float*)alloc(256 * 4);
  float* Dcnt = (float*)alloc((size_t)NN * 4);
  float* Bcnt = (float*)alloc((size_t)EE * 4);

  const long long SN = (long long)NN * CDIM;
  const long long SE = (long long)EE * CDIM;
  const unsigned GEMM_N_BLK  = (NN / 16) * (CDIM / 16) / 4;   // 25000
  const unsigned GEMM_E_BLK  = (EE / 16) * (CDIM / 16) / 4;   // 5000
  const unsigned GEMM_L_BLK  = (EE / 16) * (ODIM / 16) / 4;   // 2500
  const unsigned SCAT_BLK    = (unsigned)(((long long)NNZK * 64 + 255) / 256);

  // ---- convert weights to bf16 ----
  k_cvt_bf16<<<256, 256, 0, stream>>>(W_in,     in_proj_w, 256, 256, 256);
  k_cvt_bf16<<<256, 256, 0, stream>>>(W_e,      e_proj_w,  256, 256, 256);
  k_cvt_bf16<<<256, 256, 0, stream>>>(W_hg0,    hg0_w,     256, 256, 256);
  k_cvt_bf16<<<256, 256, 0, stream>>>(W_skip0,  skip0_w,   256, 256, 256);
  k_cvt_bf16<<<256, 256, 0, stream>>>(W_hgd0,   hgd0_w,    256, 256, 256);
  k_cvt_bf16<<<256, 256, 0, stream>>>(W_skipd0, skipd0_w,  256, 256, 256);
  k_cvt_bf16<<<256, 256, 0, stream>>>(W_fuse1,  fuse_w,        256, 256, 512);
  k_cvt_bf16<<<256, 256, 0, stream>>>(W_fuse2,  fuse_w + 256,  256, 256, 512);
  k_cvt_bf16<<<128, 256, 0, stream>>>(W_lin,    lin_w,     128, 256, 256);

  // ---- degrees ----
  k_fill<<<gs_blocks(NN), 256, 0, stream>>>(Dcnt, NN, 0.f);
  k_fill<<<gs_blocks(EE), 256, 0, stream>>>(Bcnt, EE, 0.f);
  k_degree<<<1954, 256, 0, stream>>>(row, col, Dcnt, Bcnt, NNZK);

  // ================= node path =================
  k_ln_bf16<<<NN / 8, 256, 0, stream>>>(x, in_norm_w, in_norm_b, nbf, NN, 0);
  k_gemm<<<GEMM_N_BLK, 128, 0, stream>>>(nbf, W_in, in_proj_b, fA, NN, CDIM, CDIM, FBIAS | FLEAKY);

  k_fill<<<1, 256, 0, stream>>>(cm, 256, 0.f);
  k_fill<<<1, 256, 0, stream>>>(cv, 256, 0.f);
  k_colsum<<<(NN + 255) / 256, 256, 0, stream>>>(fA, cm, NN);
  k_colvar<<<(NN + 255) / 256, 256, 0, stream>>>(fA, cm, gn0_ms, cv, NN, 1.f / NN);
  k_gn_apply<<<4096, 256, 0, stream>>>(fA, fB, nbf, cm, cv, gn0_w, gn0_b, gn0_ms, SN, 1.f / NN);

  k_gemm<<<GEMM_N_BLK, 128, 0, stream>>>(nbf, W_hg0, nullptr, fC, NN, CDIM, CDIM, 0);   // xw

  k_fill<<<4096, 256, 0, stream>>>(eA, SE, 0.f);
  k_scatter_add<<<SCAT_BLK, 256, 0, stream>>>(eA, fC, col, row, nullptr, NNZK);  // node -> hyperedge
  k_fill<<<4096, 256, 0, stream>>>(fA, SN, 0.f);
  k_scatter_add<<<SCAT_BLK, 256, 0, stream>>>(fA, eA, row, col, Bcnt, NNZK);     // (Binv*e) -> node
  k_scale_bias_leaky<<<4096, 256, 0, stream>>>(fA, Dcnt, hg0_b, SN);
  k_gemm<<<GEMM_N_BLK, 128, 0, stream>>>(nbf, W_skip0, skip0_b, fA, NN, CDIM, CDIM, FBIAS | FACC);
  // fA = x_final

  // ================= dual path =================
  k_ln_bf16<<<EE / 8, 256, 0, stream>>>(xe, e_norm_w, e_norm_b, ebf, EE, 1);
  k_gemm<<<GEMM_E_BLK, 128, 0, stream>>>(ebf, W_e, e_proj_b, eA, EE, CDIM, CDIM, FBIAS | FLEAKY);

  k_fill<<<1, 256, 0, stream>>>(cm, 256, 0.f);
  k_fill<<<1, 256, 0, stream>>>(cv, 256, 0.f);
  k_colsum<<<(EE + 255) / 256, 256, 0, stream>>>(eA, cm, EE);
  k_colvar<<<(EE + 255) / 256, 256, 0, stream>>>(eA, cm, gnd0_ms, cv, EE, 1.f / EE);
  k_gn_apply<<<4096, 256, 0, stream>>>(eA, eB, ebf, cm, cv, gnd0_w, gnd0_b, gnd0_ms, SE, 1.f / EE);

  k_gemm<<<GEMM_E_BLK, 128, 0, stream>>>(ebf, W_hgd0, nullptr, eC, EE, CDIM, CDIM, 0);  // ew

  k_fill<<<4096, 256, 0, stream>>>(fB, SN, 0.f);
  k_scatter_add<<<SCAT_BLK, 256, 0, stream>>>(fB, eC, row, col, nullptr, NNZK);  // hyperedge -> node
  k_fill<<<4096, 256, 0, stream>>>(eA, SE, 0.f);
  k_scatter_add<<<SCAT_BLK, 256, 0, stream>>>(eA, fB, col, row, Dcnt, NNZK);     // (Dinv*e) -> hyperedge
  k_scale_bias_leaky<<<4096, 256, 0, stream>>>(eA, Bcnt, hgd0_b, SE);
  k_gemm<<<GEMM_E_BLK, 128, 0, stream>>>(ebf, W_skipd0, skipd0_b, eA, EE, CDIM, CDIM, FBIAS | FACC);
  // eA = xe_final

  // ================= segment-min aggregate =================
  k_fill<<<4096, 256, 0, stream>>>(eC, SE, __builtin_inff());
  k_scatter_min<<<SCAT_BLK, 256, 0, stream>>>(eC, fA, col, row, NNZK);
  k_fixfinite<<<4096, 256, 0, stream>>>(eC, SE);

  // ================= fuse + lin =================
  k_cvt_bf16<<<4096, 256, 0, stream>>>(ebf, eC, EE, CDIM, CDIM);        // agg
  k_gemm<<<GEMM_E_BLK, 128, 0, stream>>>(ebf, W_fuse1, nullptr, eB, EE, CDIM, CDIM, 0);
  k_cvt_bf16<<<4096, 256, 0, stream>>>(ebf, eA, EE, CDIM, CDIM);        // xe_final
  k_gemm<<<GEMM_E_BLK, 128, 0, stream>>>(ebf, W_fuse2, fuse_b, eB, EE, CDIM, CDIM, FBIAS | FACC);
  k_cvt_bf16<<<4096, 256, 0, stream>>>(ebf, eB, EE, CDIM, CDIM);        // h
  k_gemm<<<GEMM_L_BLK, 128, 0, stream>>>(ebf, W_lin, lin_b, out, EE, ODIM, CDIM, FBIAS);
}
